// Model_83021717832843
// MI455X (gfx1250) — matrix operand order repaired and checked by reading, not compiled, not run on hardware
//
#include <hip/hip_runtime.h>
#include <cstdint>

typedef __attribute__((ext_vector_type(8)))  float   v8f;
typedef __attribute__((ext_vector_type(8)))  __bf16  v8bf;
typedef __attribute__((ext_vector_type(16))) __bf16  v16bf;

constexpr int kH     = 128;   // hidden dim
constexpr int kLB    = 64;    // filter steps
constexpr int kPW    = 32;    // prediction steps
constexpr int kNS    = 96;    // total steps
constexpr int kBStep = kH * kH;        // bf16 elems per packed B step (16384)
constexpr int kRows  = 128;            // batch rows per workgroup

#define WMMA_BF16(a, b, c) \
  __builtin_amdgcn_wmma_f32_16x16x32_bf16(false, (a), false, (b), (short)0, (c), false, false)

// ---- CDNA5 async memory->LDS (ASYNCcnt) helpers -----------------------------
// 16 bytes per lane, no VGPR data; LDS address = low 32 bits of generic pointer.
__device__ __forceinline__ void async_copy_b128(const void* gsrc, void* ldst) {
  asm volatile("global_load_async_to_lds_b128 %0, %1, off"
               :: "v"((unsigned)(uintptr_t)ldst), "v"(gsrc)
               : "memory");
}

template <int N>
__device__ __forceinline__ void wait_asynccnt() {
#if __has_builtin(__builtin_amdgcn_s_wait_asynccnt)
  __builtin_amdgcn_s_wait_asynccnt((short)N);
#else
  if constexpr (N == 0) asm volatile("s_wait_asynccnt 0x0" ::: "memory");
  else                  asm volatile("s_wait_asynccnt 0x8" ::: "memory");
#endif
}

__device__ __forceinline__ v16bf cvt16(float4 a0, float4 a1, float4 a2, float4 a3) {
  v16bf r;
  r[0]  = (__bf16)a0.x; r[1]  = (__bf16)a0.y; r[2]  = (__bf16)a0.z; r[3]  = (__bf16)a0.w;
  r[4]  = (__bf16)a1.x; r[5]  = (__bf16)a1.y; r[6]  = (__bf16)a1.z; r[7]  = (__bf16)a1.w;
  r[8]  = (__bf16)a2.x; r[9]  = (__bf16)a2.y; r[10] = (__bf16)a2.z; r[11] = (__bf16)a2.w;
  r[12] = (__bf16)a3.x; r[13] = (__bf16)a3.y; r[14] = (__bf16)a3.z; r[15] = (__bf16)a3.w;
  return r;
}

// A-operand (16x32 bf16) for one K-chunk from an f32 row:
// lanes 0-15: K = kb+0..15 ; lanes 16-31: K = kb+16..31
__device__ __forceinline__ v16bf load_a_f32(const float* row, int kb, int s8) {
  const float* p = row + kb;
  float4 a0 = *(const float4*)(p + 2 * s8);
  float4 a1 = *(const float4*)(p + 2 * s8 + 4);
  float4 a2 = *(const float4*)(p + 2 * s8 + 8);
  float4 a3 = *(const float4*)(p + 2 * s8 + 12);
  return cvt16(a0, a1, a2, a3);
}

__device__ __forceinline__ v16bf load_a_bf16(const __bf16* row, int kb, int s8) {
  v8bf lo = *(const v8bf*)(row + kb + 2 * s8);
  v8bf hh = *(const v8bf*)(row + kb + 2 * s8 + 8);
  return __builtin_shufflevector(lo, hh, 0, 1, 2, 3, 4, 5, 6, 7,
                                        8, 9, 10, 11, 12, 13, 14, 15);
}

// 16 consecutive f32 -> v16bf (B-operand when source row is already K-major)
__device__ __forceinline__ v16bf load_16f32(const float* p) {
  float4 a0 = *(const float4*)(p);
  float4 a1 = *(const float4*)(p + 4);
  float4 a2 = *(const float4*)(p + 8);
  float4 a3 = *(const float4*)(p + 12);
  return cvt16(a0, a1, a2, a3);
}

// ---------------------------------------------------------------------------
// Phase P: pack F_w^T into per-lane WMMA B-register order (bf16).
// B element (k,n) = F_w[t][n][k].  Layout: [t][kc][nt][lane][16 bf16],
// lanes 0-15: k = kc*32 + e ; lanes 16-31: k = kc*32 + 16 + e ; n = nt*16 + (lane&15)
// ---------------------------------------------------------------------------
__global__ __launch_bounds__(256) void pack_b_kernel(
    const float* __restrict__ Fw, __bf16* __restrict__ Bp, int total) {
  int idx = blockIdx.x * 256 + threadIdx.x;
  if (idx >= total) return;
  int e    = idx & 15;
  int lane = (idx >> 4) & 31;
  int nt   = (idx >> 9) & 7;
  int kc   = (idx >> 12) & 3;
  int t    = idx >> 14;
  int k = kc * 32 + ((lane >> 4) << 4) + e;
  int n = nt * 16 + (lane & 15);
  Bp[idx] = (__bf16)Fw[t * kBStep + n * kH + k];
}

// ---------------------------------------------------------------------------
// Phase A: batch-independent covariance recursion -> Kalman gains K_t (f32).
// One workgroup, 1024 threads (32 waves), WMMA for the two 128^3 matmuls/step.
// ---------------------------------------------------------------------------
__global__ __launch_bounds__(1024) void kalman_gain_kernel(
    const float* __restrict__ Fw, const float* __restrict__ Hw,
    const float* __restrict__ icov, const float* __restrict__ Q,
    const float* __restrict__ R, float* __restrict__ Kg) {
  __shared__ __attribute__((aligned(32))) __bf16 PbfT[kH * kH];  // P^T, bf16 (B operand)
  __shared__ __attribute__((aligned(32))) __bf16 Tbf[kH * kH];   // T = F@P, bf16 (A operand)
  __shared__ __attribute__((aligned(16))) float  Ppred[kH * kH]; // P_pred f32
  __shared__ float HwL[kH], vL[kH], wL[kH], KL[kH];
  __shared__ float SinvL;

  const int tid  = threadIdx.x;
  const int lane = tid & 31;
  const int wv   = tid >> 5;      // 0..31
  const int ln15 = lane & 15;
  const int hi   = lane >> 4;
  const int s8   = hi ? 8 : 0;
  const int h16  = hi ? 16 : 0;

  // PbfT[n][j] = init_cov[j][n]
  for (int r = 0; r < 16; ++r) {
    int idx = tid + r * 1024;
    int j = idx >> 7, n = idx & 127;
    PbfT[n * kH + j] = (__bf16)icov[j * kH + n];
  }
  __syncthreads();

  for (int t = 0; t < kLB; ++t) {
    const float* F = Fw + t * kBStep;
    // stream next step's F through L2/L0 while this step computes
    if (t + 1 < kLB) __builtin_prefetch(Fw + (t + 1) * kBStep + tid * 16, 0, 1);
    if (tid < kH) HwL[tid] = Hw[t * kH + tid];
    __syncthreads();

    // ---- T = F @ P   (B[j,k] = P[j][k] = PbfT[k][j])
    for (int tl = 0; tl < 2; ++tl) {
      const int tile = wv * 2 + tl;
      const int mt = tile >> 3, nt = tile & 7;
      const int arow = mt * 16 + ln15;
      const int ncol = nt * 16 + ln15;
      v8f acc;
      #pragma unroll
      for (int v = 0; v < 8; ++v) acc[v] = 0.0f;
      #pragma unroll
      for (int kb = 0; kb < kH; kb += 32) {
        v16bf a = load_a_f32(F + arow * kH, kb, s8);
        v16bf b = *(const v16bf*)&PbfT[ncol * kH + kb + h16];
        acc = WMMA_BF16(a, b, acc);
      }
      #pragma unroll
      for (int v = 0; v < 8; ++v)
        Tbf[(mt * 16 + v + hi * 8) * kH + nt * 16 + ln15] = (__bf16)acc[v];
    }
    __syncthreads();

    // ---- P_pred = T @ F^T + Q   (B[k,l] = F[l][k], K-contiguous in F row l)
    for (int tl = 0; tl < 2; ++tl) {
      const int tile = wv * 2 + tl;
      const int mt = tile >> 3, nt = tile & 7;
      const int arow = mt * 16 + ln15;
      const int bcol = nt * 16 + ln15;
      v8f acc;
      #pragma unroll
      for (int v = 0; v < 8; ++v)
        acc[v] = Q[(mt * 16 + v + hi * 8) * kH + nt * 16 + ln15];
      #pragma unroll
      for (int kb = 0; kb < kH; kb += 32) {
        v16bf a = load_a_bf16(Tbf + arow * kH, kb, s8);
        v16bf b = load_16f32(F + bcol * kH + kb + h16);
        acc = WMMA_BF16(a, b, acc);
      }
      #pragma unroll
      for (int v = 0; v < 8; ++v)
        Ppred[(mt * 16 + v + hi * 8) * kH + nt * 16 + ln15] = acc[v];
    }
    __syncthreads();

    // v = P_pred @ Hw
    if (tid < kH) {
      float s = 0.f;
      for (int k = 0; k < kH; ++k) s += Ppred[tid * kH + k] * HwL[k];
      vL[tid] = s;
    }
    __syncthreads();
    // S = Hw . v + R ; Sinv = 1/S   (C == 1)
    if (tid < 32) {
      float s = 0.f;
      for (int i = lane; i < kH; i += 32) s += HwL[i] * vL[i];
      #pragma unroll
      for (int off = 16; off > 0; off >>= 1) s += __shfl_xor(s, off, 32);
      if (lane == 0) SinvL = 1.0f / (s + R[0]);
    }
    __syncthreads();
    // K = v * Sinv ; w = Hw @ P_pred
    if (tid < kH) {
      float kk = vL[tid] * SinvL;
      KL[tid] = kk;
      Kg[t * kH + tid] = kk;
      float s = 0.f;
      for (int j = 0; j < kH; ++j) s += HwL[j] * Ppred[j * kH + tid];
      wL[tid] = s;
    }
    __syncthreads();
    // P_new[j][l] = P_pred[j][l] - K[j]*w[l]  -> stored transposed bf16
    for (int r = 0; r < 16; ++r) {
      int idx = tid + r * 1024;
      int j = idx >> 7, l = idx & 127;
      PbfT[l * kH + j] = (__bf16)(Ppred[j * kH + l] - KL[j] * wL[l]);
    }
    __syncthreads();
  }
}

// ---------------------------------------------------------------------------
// Phase B: per-batch state chain. 64 WGs x 256 thr; wave owns 16 batch rows.
// Per step: hs_pred = hs @ Fw^T + Fb (WMMA, 32/wave), y_pred via shuffle
// reduction, rank-1 gain update, hs round-trips through LDS (f32).
// Packed B double-buffered in LDS via global_load_async_to_lds_b128 (ASYNCcnt).
// ---------------------------------------------------------------------------
__global__ __launch_bounds__(256) void kalman_batch_kernel(
    const float* __restrict__ x, const float* __restrict__ Fb,
    const float* __restrict__ Hw, const float* __restrict__ Hb,
    const float* __restrict__ init_state, const __bf16* __restrict__ Bp,
    const float* __restrict__ Kg, float* __restrict__ out) {
  __shared__ __attribute__((aligned(16))) float  hsS[kRows * kH];   // 64KB
  __shared__ __attribute__((aligned(32))) __bf16 Bbuf[2][kBStep];   // 2x32KB
  __shared__ float FbL[kH], HwL[kH], KL[kH], yL[kRows];
  __shared__ float HbS;

  const int tid  = threadIdx.x;
  const int lane = tid & 31;
  const int wv   = tid >> 5;      // 8 waves
  const int ln15 = lane & 15;
  const int hi   = lane >> 4;
  const int s8   = hi ? 8 : 0;
  const int row0 = blockIdx.x * kRows;

  for (int r = 0; r < 64; ++r) {
    int idx = tid + r * 256;
    hsS[idx] = init_state[idx & (kH - 1)];
  }
  if (tid < kH) {
    FbL[tid] = Fb[tid];
    HwL[tid] = Hw[tid];
    KL[tid]  = Kg[tid];
    yL[tid]  = x[(row0 + tid) * kLB];
  }
  if (tid == 0) HbS = Hb[0];
  // kick off async copy of step-0 packed B into buffer 0
  {
    const char* gsrc = (const char*)Bp;
    char* ldst = (char*)&Bbuf[0][0];
    #pragma unroll
    for (int r = 0; r < 8; ++r) {
      const int off = (tid + r * 256) * 16;
      async_copy_b128(gsrc + off, ldst + off);
    }
  }

  for (int t = 0; t < kNS; ++t) {
    const int cur = t & 1;
    // issue async copy for step t+1 into the other buffer (overlaps compute)
    if (t + 1 < kNS) {
      const char* gsrc = (const char*)(Bp + (size_t)(t + 1) * kBStep);
      char* ldst = (char*)&Bbuf[cur ^ 1][0];
      #pragma unroll
      for (int r = 0; r < 8; ++r) {
        const int off = (tid + r * 256) * 16;
        async_copy_b128(gsrc + off, ldst + off);
      }
      wait_asynccnt<8>();   // oldest 8 (step t's copy) complete; newest 8 in flight
    } else {
      wait_asynccnt<0>();
    }
    __syncthreads();        // all waves' portions of Bbuf[cur] + step-t vectors visible

    // hs_pred = hs @ Fw^T + Fb  (Fb folded into accumulator init)
    v8f acc[8];
    #pragma unroll
    for (int nt = 0; nt < 8; ++nt) {
      const float fb = FbL[nt * 16 + ln15];
      #pragma unroll
      for (int v = 0; v < 8; ++v) acc[nt][v] = fb;
    }
    const int mrow = wv * 16 + ln15;
    #pragma unroll
    for (int kc = 0; kc < 4; ++kc) {
      v16bf a = load_a_f32(hsS + mrow * kH, kc * 32, s8);
      #pragma unroll
      for (int nt = 0; nt < 8; ++nt) {
        v16bf b = *(const v16bf*)&Bbuf[cur][((kc * 8 + nt) * 32 + lane) * 16];
        acc[nt] = WMMA_BF16(a, b, acc[nt]);
      }
    }

    // y_pred[row] = hs_pred . Hw (+Hb) : partials over N then 16-lane butterfly
    float part[8];
    #pragma unroll
    for (int v = 0; v < 8; ++v) part[v] = 0.f;
    #pragma unroll
    for (int nt = 0; nt < 8; ++nt) {
      const float hw = HwL[nt * 16 + ln15];
      #pragma unroll
      for (int v = 0; v < 8; ++v) part[v] += acc[nt][v] * hw;
    }
    #pragma unroll
    for (int off = 1; off < 16; off <<= 1) {
      #pragma unroll
      for (int v = 0; v < 8; ++v) part[v] += __shfl_xor(part[v], off, 32);
    }

    if (t < kLB) {  // filter: hs += K * (y - y_pred)
      float rsd[8];
      #pragma unroll
      for (int v = 0; v < 8; ++v)
        rsd[v] = yL[wv * 16 + v + hi * 8] - (part[v] + HbS);
      #pragma unroll
      for (int nt = 0; nt < 8; ++nt) {
        const float kk = KL[nt * 16 + ln15];
        #pragma unroll
        for (int v = 0; v < 8; ++v) acc[nt][v] += kk * rsd[v];
      }
    } else {        // prediction: emit em = y_pred
      if (ln15 == 0) {
        #pragma unroll
        for (int v = 0; v < 8; ++v)
          out[(row0 + wv * 16 + v + hi * 8) * kPW + (t - kLB)] = part[v] + HbS;
      }
    }

    // store hs (C-layout regs -> row-major LDS; rows are wave-private)
    #pragma unroll
    for (int nt = 0; nt < 8; ++nt) {
      #pragma unroll
      for (int v = 0; v < 8; ++v)
        hsS[(wv * 16 + v + hi * 8) * kH + nt * 16 + ln15] = acc[nt][v];
    }

    __syncthreads();  // everyone done reading Bbuf[cur] / step-t vectors
    if (t + 1 < kNS) {
      const int tn = t + 1;
      if (tid < kH) {
        FbL[tid] = Fb[tn * kH + tid];
        HwL[tid] = Hw[tn * kH + tid];
        if (tn < kLB) {
          KL[tid] = Kg[tn * kH + tid];
          yL[tid] = x[(row0 + tid) * kLB + tn];
        }
      }
      if (tid == 0) HbS = Hb[tn];
    }
  }
}

extern "C" void kernel_launch(void* const* d_in, const int* in_sizes, int n_in,
                              void* d_out, int out_size, void* d_ws, size_t ws_size,
                              hipStream_t stream) {
  (void)in_sizes; (void)n_in; (void)out_size; (void)ws_size;
  const float* x  = (const float*)d_in[0];
  const float* Fw = (const float*)d_in[1];
  const float* Fb = (const float*)d_in[2];
  const float* Hw = (const float*)d_in[3];
  const float* Hb = (const float*)d_in[4];
  const float* s0 = (const float*)d_in[5];
  const float* P0 = (const float*)d_in[6];
  const float* Q  = (const float*)d_in[7];
  const float* R  = (const float*)d_in[8];
  float* out = (float*)d_out;

  __bf16* Bpack = (__bf16*)d_ws;                                          // 3 MB
  float*  Kg    = (float*)((char*)d_ws + (size_t)kNS * kBStep * sizeof(__bf16));

  const int total = kNS * kBStep;
  pack_b_kernel<<<(total + 255) / 256, 256, 0, stream>>>(Fw, Bpack, total);
  kalman_gain_kernel<<<1, 1024, 0, stream>>>(Fw, Hw, P0, Q, R, Kg);
  kalman_batch_kernel<<<8192 / kRows, 256, 0, stream>>>(x, Fb, Hw, Hb, s0,
                                                        Bpack, Kg, out);
}
